// NPHierarchicalTransitionPrior_670014898813
// MI455X (gfx1250) — matrix operand order repaired and checked by reading, not compiled, not run on hardware
//
#include <hip/hip_runtime.h>
#include <hip/hip_bf16.h>

typedef __attribute__((ext_vector_type(16))) _Float16 v16h;
typedef __attribute__((ext_vector_type(8)))  _Float16 v8h;
typedef __attribute__((ext_vector_type(8)))  float    v8f;
typedef __attribute__((ext_vector_type(4)))  int      v4i;

#define WMMA_F16(a, b, c) \
  __builtin_amdgcn_wmma_f32_16x16x32_f16(false, (a), false, (b), (short)0, (c), false, false)

// ---- problem constants ---------------------------------------------------
constexpr int    kB      = 256;
constexpr int    kT      = 513;
constexpr int    kD      = 16;
constexpr int    kN      = kB * (kT - 1);                // 131072 samples
constexpr size_t RES_OFF = 0;                            // residuals: N*16 floats
constexpr size_t SUM_OFF = (size_t)kN * 16;              // 2097152
constexpr size_t H0_OFF  = SUM_OFF + kB;                 // 2097408, hist0: 8*N*8
constexpr size_t H1_OFF  = H0_OFF + (size_t)8 * kN * 8;  // 10486016, hist1: 8*N*16
constexpr int    NWAVES  = 8;                            // 16-sample tiles per block

// ---- A-fragment loader, row-major LDS [M][K] -----------------------------
// 16-bit A 16x32: e=0..15 -> K=16*half+e
__device__ __forceinline__ v16h ldfragA(const _Float16* W, int stride, int m, int ks, int half) {
  const _Float16* p  = W + m * stride + ks * 32 + 16 * half;
  v8h lo = *(const v8h*)p;          // ds_load_b128
  v8h hi = *(const v8h*)(p + 8);    // ds_load_b128
  v16h f;
#pragma unroll
  for (int i = 0; i < 8; ++i) { f[i] = lo[i]; f[i + 8] = hi[i]; }
  return f;
}

// ---- register-only cross-half exchange helpers (wave32) ------------------
__device__ __forceinline__ v4i shflx16(v4i x) {
  v4i r;
#pragma unroll
  for (int i = 0; i < 4; ++i) r[i] = __shfl_xor(x[i], 16);
  return r;
}
__device__ __forceinline__ v16h cat16(v8h lo, v8h hi) {
  v16h f;
#pragma unroll
  for (int i = 0; i < 8; ++i) { f[i] = lo[i]; f[i + 8] = hi[i]; }
  return f;
}
// Build B fragment (K = ks*32 + 16*half + e) from the per-lane packed C rows:
// a0 = f16(acc[2ks]) rows mt*16+8*half+r, a1 = f16(acc[2ks+1]).
// lane keeps acc[2ks+half], receives partner's acc[2ks+half] via xor-16.
__device__ __forceinline__ v16h xchgB(v8h a0, v8h a1, int half) {
  v4i q0 = __builtin_bit_cast(v4i, a0);
  v4i q1 = __builtin_bit_cast(v4i, a1);
  v4i send = half ? q0 : q1;
  v4i recv = shflx16(send);
  v4i own  = half ? q1 : q0;
  v4i lo   = half ? recv : own;   // rows mt*16+0..7   (from half0 lane)
  v4i hi   = half ? own  : recv;  // rows mt*16+8..15  (from half1 lane)
  return cat16(__builtin_bit_cast(v8h, lo), __builtin_bit_cast(v8h, hi));
}

// One block = 256 threads = 8 waves; each wave owns a 16-sample tile.
// blockIdx.y = subnet s in [0,16): layer = s>>3, n = s&7.
__global__ __launch_bounds__(256)
void nphier_mlp_fwdgrad(const float* __restrict__ x,
                        const float* __restrict__ l0W1, const float* __restrict__ l0b1,
                        const float* __restrict__ l0W2, const float* __restrict__ l0b2,
                        const float* __restrict__ l0W3, const float* __restrict__ l0b3,
                        const float* __restrict__ l1W1, const float* __restrict__ l1b1,
                        const float* __restrict__ l1W2, const float* __restrict__ l1b2,
                        const float* __restrict__ l1W3, const float* __restrict__ l1b3,
                        float* __restrict__ out) {
  const int s     = blockIdx.y;
  const int layer = s >> 3;
  const int nsub  = s & 7;
  const int idim  = layer ? 17 : 9;

  const float* pW1 = layer ? l1W1 : l0W1;
  const float* pB1 = layer ? l1b1 : l0b1;
  const float* pW2 = layer ? l1W2 : l0W2;
  const float* pB2 = layer ? l1b2 : l0b2;
  const float* pW3 = layer ? l1W3 : l0W3;
  const float* pB3 = layer ? l1b3 : l0b3;

  // ---- LDS: f16 weights only (+ transposed copies for backward) ----------
  __shared__ _Float16 sW1 [64 * 32];   // W1 extended with bias column at k=idim
  __shared__ _Float16 sW1T[32 * 64];
  __shared__ _Float16 sW2 [64 * 64];
  __shared__ _Float16 sW2T[64 * 64];
  __shared__ float    sB2[64];
  __shared__ float    sW3v[64];
  __shared__ float    sB3;

  const int tid = threadIdx.x;
  for (int i = tid; i < 64 * 32; i += 256) {
    int m = i >> 5, k = i & 31;
    float v = (k < idim) ? pW1[(nsub * 64 + m) * idim + k]
            : (k == idim) ? pB1[nsub * 64 + m] : 0.f;
    _Float16 h = (_Float16)v;
    sW1[m * 32 + k]  = h;
    sW1T[k * 64 + m] = h;
  }
  for (int i = tid; i < 64 * 64; i += 256) {
    int m = i >> 6, k = i & 63;
    _Float16 h = (_Float16)pW2[(nsub * 64 + m) * 64 + k];
    sW2[m * 64 + k]  = h;
    sW2T[k * 64 + m] = h;
  }
  if (tid < 64) { sB2[tid] = pB2[nsub * 64 + tid]; sW3v[tid] = pW3[nsub * 64 + tid]; }
  if (tid == 0) sB3 = pB3[nsub];
  __syncthreads();

  const int wave = tid >> 5;
  const int lane = tid & 31;
  const int ncol = lane & 15;
  const int half = lane >> 4;
  const int tilebase = (blockIdx.x * NWAVES + wave) * 16;

  // ---- build the 32x16 input tile V in registers (bias ones-row @ idim) --
  v8h p0, p1, p2, p3;  // rows 0..7 / 8..15 / 16..23 / 24..31 of column ncol
  if (half == 0) {
    const int idx = tilebase + ncol;
    const int b = idx >> 9, t = idx & 511;
    const float4* px = (const float4*)(x + (size_t)(b * kT + t) * kD);
    float v[32];
#pragma unroll
    for (int k = 0; k < 32; ++k) v[k] = 0.f;
    if (layer == 0) {
      float4 y2 = px[2], y3 = px[3];                       // yy[8..15]
      v[0] = y2.x; v[1] = y2.y; v[2] = y2.z; v[3] = y2.w;
      v[4] = y3.x; v[5] = y3.y; v[6] = y3.z; v[7] = y3.w;
      v[8] = x[(size_t)(b * kT + t + 1) * kD + 8 + nsub];  // xcol
      v[9] = 1.f;                                          // bias row
    } else {
      float4 y0 = px[0], y1 = px[1];                       // yy[0..7]
      float4 x2 = px[6], x3 = px[7];                       // xx[8..15]
      v[0]  = y0.x; v[1]  = y0.y; v[2]  = y0.z; v[3]  = y0.w;
      v[4]  = y1.x; v[5]  = y1.y; v[6]  = y1.z; v[7]  = y1.w;
      v[8]  = x2.x; v[9]  = x2.y; v[10] = x2.z; v[11] = x2.w;
      v[12] = x3.x; v[13] = x3.y; v[14] = x3.z; v[15] = x3.w;
      v[16] = x[(size_t)(b * kT + t + 1) * kD + nsub];     // xcol
      v[17] = 1.f;                                         // bias row
    }
#pragma unroll
    for (int i = 0; i < 8; ++i) {
      p0[i] = (_Float16)v[i];      p1[i] = (_Float16)v[8 + i];
      p2[i] = (_Float16)v[16 + i]; p3[i] = (_Float16)v[24 + i];
    }
  }
  // half0 lanes own rows 0..15; half1 lanes get rows 16..31 via xor-16
  v16h bV;
  {
    v4i r2 = shflx16(__builtin_bit_cast(v4i, p2));
    v4i r3 = shflx16(__builtin_bit_cast(v4i, p3));
    v4i lo = half ? r2 : __builtin_bit_cast(v4i, p0);
    v4i hi = half ? r3 : __builtin_bit_cast(v4i, p1);
    bV = cat16(__builtin_bit_cast(v8h, lo), __builtin_bit_cast(v8h, hi));
  }

  const v8f vzero = {0.f, 0.f, 0.f, 0.f, 0.f, 0.f, 0.f, 0.f};

  // ---- GEMM1: Z1(64x16) = W1ext(64x32) x V(32x16), bias folded in --------
  v8f acc1[4];
#pragma unroll
  for (int mt = 0; mt < 4; ++mt) {
    v16h a = ldfragA(sW1, 32, mt * 16 + ncol, 0, half);
    acc1[mt] = WMMA_F16(a, bV, vzero);
  }
  // leaky -> H1 kept in acc1; pack f16 rows for the register exchange
  v8h hk[4];
#pragma unroll
  for (int mt = 0; mt < 4; ++mt) {
#pragma unroll
    for (int r = 0; r < 8; ++r) {
      float z = acc1[mt][r];
      float h = z >= 0.f ? z : 0.01f * z;
      acc1[mt][r] = h;
      hk[mt][r] = (_Float16)h;
    }
  }

  // ---- GEMM2: Z2(64x16) = W2(64x64) x H1(64x16) ---------------------------
  v8f acc2[4] = {vzero, vzero, vzero, vzero};
#pragma unroll
  for (int ks = 0; ks < 2; ++ks) {
    v16h bh = xchgB(hk[2 * ks], hk[2 * ks + 1], half);
#pragma unroll
    for (int mt = 0; mt < 4; ++mt) {
      v16h a = ldfragA(sW2, 64, mt * 16 + ncol, ks, half);
      acc2[mt] = WMMA_F16(a, bh, acc2[mt]);
    }
  }
  // bias + leaky + W3 dot (output head); pack dZ2 = W3 * leaky'(Z2)
  float pdot = 0.f;
  v8h dk[4];
#pragma unroll
  for (int mt = 0; mt < 4; ++mt) {
#pragma unroll
    for (int r = 0; r < 8; ++r) {
      int row = mt * 16 + 8 * half + r;
      float w3 = sW3v[row];
      float z = acc2[mt][r] + sB2[row];
      float h = z >= 0.f ? z : 0.01f * z;
      pdot += w3 * h;
      dk[mt][r] = (_Float16)((z >= 0.f ? 1.f : 0.01f) * w3);
    }
  }
  float ptot = pdot + __shfl_xor(pdot, 16);
  if (half == 0)
    out[RES_OFF + (size_t)(tilebase + ncol) * 16 + s] = ptot + sB3;

  // ---- GEMM3: dH1(64x16) = W2^T x dZ2 -------------------------------------
  v8f accB[4] = {vzero, vzero, vzero, vzero};
#pragma unroll
  for (int ks = 0; ks < 2; ++ks) {
    v16h bh = xchgB(dk[2 * ks], dk[2 * ks + 1], half);
#pragma unroll
    for (int mt = 0; mt < 4; ++mt) {
      v16h a = ldfragA(sW2T, 64, mt * 16 + ncol, ks, half);
      accB[mt] = WMMA_F16(a, bh, accB[mt]);
    }
  }
  // dZ1 = dH1 * leaky'(Z1) (sign of H1 == sign of Z1)
  v8h gk[4];
#pragma unroll
  for (int mt = 0; mt < 4; ++mt) {
#pragma unroll
    for (int r = 0; r < 8; ++r) {
      float d = accB[mt][r] * (acc1[mt][r] >= 0.f ? 1.f : 0.01f);
      gk[mt][r] = (_Float16)d;
    }
  }

  // ---- GEMM4: grad(32x16) = W1ext^T x dZ1 ---------------------------------
  v8f accG[2] = {vzero, vzero};
#pragma unroll
  for (int ks = 0; ks < 2; ++ks) {
    v16h bh = xchgB(gk[2 * ks], gk[2 * ks + 1], half);
#pragma unroll
    for (int mt = 0; mt < 2; ++mt) {
      v16h a = ldfragA(sW1T, 64, mt * 16 + ncol, ks, half);
      accG[mt] = WMMA_F16(a, bh, accG[mt]);
    }
  }

  // ---- outputs: hist + logdet ---------------------------------------------
  const size_t idx = (size_t)tilebase + ncol;
  float lg = 0.f;
  if (layer == 0) {
    if (half == 0) {  // rows 0..7 = grad[:-1]
      float* hp = out + H0_OFF + ((size_t)nsub * kN + idx) * 8;
      ((float4*)hp)[0] = make_float4(accG[0][0], accG[0][1], accG[0][2], accG[0][3]);
      ((float4*)hp)[1] = make_float4(accG[0][4], accG[0][5], accG[0][6], accG[0][7]);
    } else {          // row 8 = grad[-1] (xcol derivative)
      lg = logf(fabsf(accG[0][0]));
    }
  } else {
    float* hp = out + H1_OFF + ((size_t)nsub * kN + idx) * 16 + 8 * half;
    ((float4*)hp)[0] = make_float4(accG[0][0], accG[0][1], accG[0][2], accG[0][3]);
    ((float4*)hp)[1] = make_float4(accG[0][4], accG[0][5], accG[0][6], accG[0][7]);
    if (half == 0)    // row 16 = grad[-1]
      lg = logf(fabsf(accG[1][0]));
  }
  // wave reduction -> one atomic per wave (all 16 samples share batch index)
#pragma unroll
  for (int o = 16; o >= 1; o >>= 1) lg += __shfl_xor(lg, o);
  if (lane == 0)
    atomicAdd(out + SUM_OFF + (tilebase >> 9), lg);
}

extern "C" void kernel_launch(void* const* d_in, const int* in_sizes, int n_in,
                              void* d_out, int out_size, void* d_ws, size_t ws_size,
                              hipStream_t stream) {
  (void)in_sizes; (void)n_in; (void)d_ws; (void)ws_size; (void)out_size;
  const float* x    = (const float*)d_in[0];
  // d_in[1] = alphas (unused by the reference outputs)
  const float* l0W1 = (const float*)d_in[2];
  const float* l0b1 = (const float*)d_in[3];
  const float* l0W2 = (const float*)d_in[4];
  const float* l0b2 = (const float*)d_in[5];
  const float* l0W3 = (const float*)d_in[6];
  const float* l0b3 = (const float*)d_in[7];
  const float* l1W1 = (const float*)d_in[8];
  const float* l1b1 = (const float*)d_in[9];
  const float* l1W2 = (const float*)d_in[10];
  const float* l1b2 = (const float*)d_in[11];
  const float* l1W3 = (const float*)d_in[12];
  const float* l1b3 = (const float*)d_in[13];
  float* out = (float*)d_out;

  // zero the sumlog accumulator region (atomics add into it)
  hipMemsetAsync(out + SUM_OFF, 0, kB * sizeof(float), stream);

  dim3 grid(kN / (16 * NWAVES), 16, 1);  // (1024, 16): sample tiles x subnets
  dim3 block(256, 1, 1);
  nphier_mlp_fwdgrad<<<grid, block, 0, stream>>>(
      x, l0W1, l0b1, l0W2, l0b2, l0W3, l0b3,
         l1W1, l1b1, l1W2, l1b2, l1W3, l1b3, out);
}
